// BidiAttention_51788715655445
// MI455X (gfx1250) — matrix operand order repaired and checked by reading, not compiled, not run on hardware
//
#include <hip/hip_runtime.h>

typedef __attribute__((ext_vector_type(16))) _Float16 v16h;
typedef __attribute__((ext_vector_type(8)))  _Float16 v8h;
typedef __attribute__((ext_vector_type(8)))  float    v8f;
typedef __attribute__((ext_vector_type(4)))  float    v4f;

#define HIDDEN 768
#define NHEADS 12
#define HDIM   64
#define BATCH  4
#define SEQ    2048

union F16x16 { v16h v; v8h h[2]; };

static __device__ inline v8f vzero8() {
  v8f z = {0.f,0.f,0.f,0.f,0.f,0.f,0.f,0.f};
  return z;
}

static __device__ inline v8f wmma_f16(v16h a, v16h b, v8f c) {
  // D = A(16x32 f16) x B(32x16 f16) + C(16x16 f32)
  return __builtin_amdgcn_wmma_f32_16x16x32_f16(
      /*neg_a=*/false, a, /*neg_b=*/false, b,
      /*c_mod=*/(short)0, c, /*reuse_a=*/false, /*reuse_b=*/false);
}

// ---------------------------------------------------------------------------
// W [K][F] fp32 -> Wt [F][K] f16   (makes B-fragment loads contiguous in K)
// ---------------------------------------------------------------------------
__global__ void wt_transpose(const float* __restrict__ W,
                             _Float16* __restrict__ Wt, int K, int F) {
  int idx = blockIdx.x * 256 + threadIdx.x;
  if (idx >= K * F) return;
  int f = idx / K, k = idx - f * K;
  Wt[idx] = (_Float16)W[(long)k * F + f];
}

// ---------------------------------------------------------------------------
// Projection GEMM: out[r, f0+n] = X[r,:] . Wt[f,:] + bias[f]
//   X  : fp32 row-major [BATCH*SEQ][Kdim]
//   Wt : f16  [768][Kdim]
//   out_rm : f16 [b][h][n][64]   (row-major per head; A / S-phase-B operand)
//   out_t  : f16 [b][h][64][n]   (transposed per head; PV-phase-B operand)
// One wave per 16x16 output tile; K accumulated 32 at a time via WMMA.
// ---------------------------------------------------------------------------
__global__ void __launch_bounds__(32)
proj_gemm(const float* __restrict__ X, const _Float16* __restrict__ Wt,
          const float* __restrict__ bias, _Float16* __restrict__ out_rm,
          _Float16* __restrict__ out_t, int Kdim) {
  const int lane = threadIdx.x;
  const int n = lane & 15, half = lane >> 4;
  const int r0 = blockIdx.x * 16;   // token tile
  const int f0 = blockIdx.y * 16;   // output-feature tile
  const long rrow = (long)(r0 + n) * Kdim;   // A row for this lane (m = lane&15)

  v8f acc = vzero8();
  for (int kk = 0; kk < Kdim; kk += 32) {
    // A-fragment: lane holds K run {half*16..+15}
    v4f x0 = *(const v4f*)(X + rrow + kk + half * 16);
    v4f x1 = *(const v4f*)(X + rrow + kk + half * 16 + 4);
    v4f x2 = *(const v4f*)(X + rrow + kk + half * 16 + 8);
    v4f x3 = *(const v4f*)(X + rrow + kk + half * 16 + 12);
    F16x16 a;
#pragma unroll
    for (int i = 0; i < 4; ++i) {
      a.v[i]      = (_Float16)x0[i];
      a.v[4 + i]  = (_Float16)x1[i];
      a.v[8 + i]  = (_Float16)x2[i];
      a.v[12 + i] = (_Float16)x3[i];
    }
    // B-fragment: lane holds col f0+n, K = kk + half*16 + e (contiguous in Wt)
    F16x16 bfr;
    const _Float16* wp = Wt + (long)(f0 + n) * Kdim + kk + half * 16;
    bfr.h[0] = *(const v8h*)wp;
    bfr.h[1] = *(const v8h*)(wp + 8);
    acc = wmma_f16(a.v, bfr.v, acc);
  }

  const float bv = bias[f0 + n];
#pragma unroll
  for (int v = 0; v < 8; ++v) acc[v] += bv;

  const int hh = f0 >> 6;     // head
  const int d0 = f0 & 63;     // dim within head
  const int bidx = r0 >> 11;  // batch (SEQ = 2048)
  const int nn0 = r0 & 2047;

  if (out_rm) {  // strided b16 stores: D row = v + 8*half, col = n
#pragma unroll
    for (int v = 0; v < 8; ++v) {
      const int row = v + 8 * half;
      const long addr =
          (((long)bidx * NHEADS + hh) * SEQ + nn0 + row) * HDIM + d0 + n;
      out_rm[addr] = (_Float16)acc[v];
    }
  }
  if (out_t) {   // lane's 8 rows are consecutive tokens -> one b128 store
    v8h t;
#pragma unroll
    for (int v = 0; v < 8; ++v) t[v] = (_Float16)acc[v];
    const long addr =
        (((long)bidx * NHEADS + hh) * HDIM + d0 + n) * SEQ + nn0 + 8 * half;
    *(v8h*)(out_t + addr) = t;
  }
}

// ---------------------------------------------------------------------------
// Flash attention, one 16-row query tile per wave.
//   Arm : "query" rows,  f16 [b][h][N][64]   (A operand)
//   Brm : "key"   rows,  f16 [b][h][N][64]   (B operand of S = A.Brm^T)
//   Vt  : "value" cols,  f16 [b][h][64][N]   (B operand of P.V)
//   maskK : additive mask over the key axis, fp32 [b][N]
//   out : fp32 [b][N][768]
// Direction 1: (Q, K, V) -> qc.  Direction 2: (K, Q, Q) -> vc.
// ---------------------------------------------------------------------------
__global__ void __launch_bounds__(32)
attn_flash(const _Float16* __restrict__ Arm, const _Float16* __restrict__ Brm,
           const _Float16* __restrict__ Vt, const float* __restrict__ maskK,
           float* __restrict__ out) {
  __shared__ alignas(16) _Float16 Pl[16 * 40];  // P tile scratch, padded rows
  const int lane = threadIdx.x;
  const int n = lane & 15, half = lane >> 4;
  const int q0 = blockIdx.x * 16;
  const int h = blockIdx.y, b = blockIdx.z;
  const long bh = (long)(b * NHEADS + h);
  const _Float16* Aq = Arm + bh * SEQ * HDIM;
  const _Float16* Kb = Brm + bh * SEQ * HDIM;
  const _Float16* Vb = Vt + bh * HDIM * SEQ;

  // Q A-fragments (16x32 over d), loaded once, reused for all key tiles.
  F16x16 aq[2];
#pragma unroll
  for (int c = 0; c < 2; ++c) {
    const _Float16* p = Aq + (long)(q0 + n) * HDIM + c * 32 + half * 16;
    aq[c].h[0] = *(const v8h*)p;
    aq[c].h[1] = *(const v8h*)(p + 8);
  }

  v8f acc[4];  // qc tile 16x64 as four 16x16 D tiles
#pragma unroll
  for (int dc = 0; dc < 4; ++dc) acc[dc] = vzero8();
  float mrun[8], lrun[8];  // per-row running max/sum; row = v + 8*half
#pragma unroll
  for (int v = 0; v < 8; ++v) { mrun[v] = -1e30f; lrun[v] = 0.f; }

  const float scale = 0.125f;  // 1/sqrt(64)

  for (int kj = 0; kj < SEQ; kj += 32) {
    // ---- S tiles: two 16x16 tiles covering keys kj..kj+31 ----
    v8f s[2];
#pragma unroll
    for (int jt = 0; jt < 2; ++jt) {
      v8f c0 = vzero8();
#pragma unroll
      for (int c = 0; c < 2; ++c) {
        F16x16 bk;  // B[k=d][col=key]: lane = key kj+16*jt+n, d contiguous
        const _Float16* p =
            Kb + (long)(kj + jt * 16 + n) * HDIM + c * 32 + half * 16;
        bk.h[0] = *(const v8h*)p;
        bk.h[1] = *(const v8h*)(p + 8);
        c0 = wmma_f16(aq[c].v, bk.v, c0);
      }
      const float mk = maskK[(long)b * SEQ + kj + jt * 16 + n];
#pragma unroll
      for (int v = 0; v < 8; ++v) s[jt][v] = c0[v] * scale + mk;
    }

    // ---- online softmax over these 32 keys ----
#pragma unroll
    for (int v = 0; v < 8; ++v) {
      float t = fmaxf(s[0][v], s[1][v]);
      t = fmaxf(t, __shfl_xor(t, 1, 32));
      t = fmaxf(t, __shfl_xor(t, 2, 32));
      t = fmaxf(t, __shfl_xor(t, 4, 32));
      t = fmaxf(t, __shfl_xor(t, 8, 32));  // reduce within each 16-lane half
      const float mnew = fmaxf(mrun[v], t);
      const float corr = __expf(mrun[v] - mnew);
      const float p0 = __expf(s[0][v] - mnew);
      const float p1 = __expf(s[1][v] - mnew);
      float rs = p0 + p1;
      rs += __shfl_xor(rs, 1, 32);
      rs += __shfl_xor(rs, 2, 32);
      rs += __shfl_xor(rs, 4, 32);
      rs += __shfl_xor(rs, 8, 32);
      lrun[v] = lrun[v] * corr + rs;
      mrun[v] = mnew;
      acc[0][v] *= corr; acc[1][v] *= corr;
      acc[2][v] *= corr; acc[3][v] *= corr;
      const int row = v + 8 * half;  // D-layout row owned by this lane/VGPR
      Pl[row * 40 + n]      = (_Float16)p0;
      Pl[row * 40 + 16 + n] = (_Float16)p1;
    }

    // D-layout -> A-layout transpose of P via LDS
    asm volatile("s_wait_dscnt 0x0" ::: "memory");
    F16x16 pa;
    pa.h[0] = *(const v8h*)(Pl + n * 40 + half * 16);
    pa.h[1] = *(const v8h*)(Pl + n * 40 + half * 16 + 8);

    // ---- acc += P(16x32) x V(32x64), four d-chunks ----
#pragma unroll
    for (int dc = 0; dc < 4; ++dc) {
      F16x16 bv;  // B[k=key][col=d]: Vt gives key contiguous per lane
      const _Float16* p = Vb + (long)(dc * 16 + n) * SEQ + kj + half * 16;
      bv.h[0] = *(const v8h*)p;
      bv.h[1] = *(const v8h*)(p + 8);
      acc[dc] = wmma_f16(pa.v, bv.v, acc[dc]);
    }
    if (kj + 32 < SEQ)  // hint next key tile toward the caches
      __builtin_prefetch(Kb + (long)(kj + 32 + n) * HDIM, 0, 1);
  }

  // ---- normalize and store: out[b][q][h*64 + d] ----
#pragma unroll
  for (int v = 0; v < 8; ++v) {
    const float inv = 1.f / lrun[v];
    const int row = v + 8 * half;
    const long ob = ((long)b * SEQ + q0 + row) * HIDDEN + (long)h * HDIM;
#pragma unroll
    for (int dc = 0; dc < 4; ++dc) out[ob + dc * 16 + n] = acc[dc][v] * inv;
  }
}

// ---------------------------------------------------------------------------
extern "C" void kernel_launch(void* const* d_in, const int* in_sizes, int n_in,
                              void* d_out, int out_size, void* d_ws,
                              size_t ws_size, hipStream_t stream) {
  (void)in_sizes; (void)n_in; (void)out_size; (void)ws_size;
  const float* query = (const float*)d_in[0];
  const float* key   = (const float*)d_in[1];
  const float* value = (const float*)d_in[2];
  const float* vmask = (const float*)d_in[3];  // [B,1,1,NK]
  const float* qmask = (const float*)d_in[4];  // [B,1,1,NQ]
  const float* Wq = (const float*)d_in[5];
  const float* bq = (const float*)d_in[6];
  const float* Wk = (const float*)d_in[7];
  const float* bk = (const float*)d_in[8];
  const float* Wv = (const float*)d_in[9];
  const float* bv = (const float*)d_in[10];

  char* ws = (char*)d_ws;
  _Float16* WtQ = (_Float16*)ws; ws += (size_t)768 * 768 * 2;
  _Float16* WtK = (_Float16*)ws; ws += (size_t)1536 * 768 * 2;
  _Float16* WtV = (_Float16*)ws; ws += (size_t)768 * 768 * 2;
  const size_t he = (size_t)BATCH * NHEADS * SEQ * HDIM;
  _Float16* Qrm = (_Float16*)ws; ws += he * 2;
  _Float16* Qt  = (_Float16*)ws; ws += he * 2;
  _Float16* Krm = (_Float16*)ws; ws += he * 2;
  _Float16* Vtr = (_Float16*)ws; ws += he * 2;

  wt_transpose<<<(768 * 768 + 255) / 256, 256, 0, stream>>>(Wq, WtQ, 768, 768);
  wt_transpose<<<(1536 * 768 + 255) / 256, 256, 0, stream>>>(Wk, WtK, 1536, 768);
  wt_transpose<<<(768 * 768 + 255) / 256, 256, 0, stream>>>(Wv, WtV, 768, 768);

  dim3 gp(BATCH * SEQ / 16, HIDDEN / 16, 1);
  proj_gemm<<<gp, 32, 0, stream>>>(query, WtQ, bq, Qrm, Qt, 768);
  proj_gemm<<<gp, 32, 0, stream>>>(key,   WtK, bk, Krm, nullptr, 1536);
  proj_gemm<<<gp, 32, 0, stream>>>(value, WtV, bv, nullptr, Vtr, 768);

  dim3 ga(SEQ / 16, NHEADS, BATCH);
  float* qc = (float*)d_out;
  float* vc = qc + (size_t)BATCH * SEQ * HIDDEN;
  // direction 1: softmax over keys, attend V
  attn_flash<<<ga, 32, 0, stream>>>(Qrm, Krm, Vtr, vmask, qc);
  // direction 2: transposed scores == flash with (Q,K,V) -> (K,Q,Q)
  attn_flash<<<ga, 32, 0, stream>>>(Krm, Qrm, Qt, qmask, vc);
}
